// GraphRec_37091337568687
// MI455X (gfx1250) — matrix operand order repaired and checked by reading, not compiled, not run on hardware
//
#include <hip/hip_runtime.h>

// ============================================================================
// GraphRec forward for MI455X (gfx1250, wave32, WMMA).
// All GEMMs run through v_wmma_f32_16x16x32_bf16 (f32 accumulate).
//   A fragments: built on the fly from f32 data (per ISA 7.12.2, lane holds
//     row M=lane&15, K = kb..kb+7 and kb+16..kb+23, kb=k0+8*(lane>>4)).
//   B fragments: pre-packed bf16, pack[(s*64+n)*16 + j] = W[16s+j][n], so a
//     lane's 16 K-values for its column are 16B halves at +8*(lane>>4) of chunks s, s+1.
// The packed form of Uh doubles as the A^T fragment source for C^T = Uh^T @ R.
// k_big_rowgather stages its gathered R/S rows into LDS with double-buffered
// GLOBAL_LOAD_ASYNC_TO_LDS_B128 (ASYNCcnt-tracked, coalesced 128B runs).
// ============================================================================

#define DEV __device__ __forceinline__

typedef float          v8f    __attribute__((ext_vector_type(8)));
typedef __bf16         v16bf  __attribute__((ext_vector_type(16)));

DEV v8f wmma_bf16(v16bf a, v16bf b, v8f c) {
  return __builtin_amdgcn_wmma_f32_16x16x32_bf16(false, a, false, b,
                                                 (short)0, c, false, false);
}

struct BF32B { uint4 lo; uint4 hi; };   // 32 bytes == one v16bf fragment

DEV v16bf load_bfrag(const unsigned short* p) {         // contiguous 32B
  BF32B r; r.lo = *(const uint4*)p; r.hi = *(const uint4*)(p + 8);
  return __builtin_bit_cast(v16bf, r);
}

DEV v16bf load_bfrag2(const unsigned short* p0, const unsigned short* p1) {
  BF32B r; r.lo = *(const uint4*)p0; r.hi = *(const uint4*)p1;
  return __builtin_bit_cast(v16bf, r);
}

// A fragment from f32 (8 + 8 consecutive floats at p, p+16), scaled -> bf16.
// Native casts lower to hardware v_cvt bf16 ops on gfx1250.
DEV v16bf build_afrag(const float* p, float sc) {
  v8f lo = *(const v8f*)(p);
  v8f hi = *(const v8f*)(p + 16);
  v16bf t;
#pragma unroll
  for (int j = 0; j < 8; ++j) {
    t[j]     = (__bf16)(lo[j] * sc);
    t[j + 8] = (__bf16)(hi[j] * sc);
  }
  return t;
}

// ---------------------------------------------------------------------------
// out[M x 64] = relu( s1 .* (X1[g1] @ W1p)  +  (X2[g2] @ W2p)  + bias )
// 16 rows per wave, 4 waves per block, grid = M/64.
// ---------------------------------------------------------------------------
__global__ __launch_bounds__(128) void k_small_gemm(
    const float* __restrict__ X1, const int* __restrict__ idx1,
    const float* __restrict__ scale1, const unsigned short* __restrict__ W1p,
    const float* __restrict__ X2, const int* __restrict__ idx2,
    const unsigned short* __restrict__ W2p,
    const float* __restrict__ bias, float* __restrict__ out)
{
  const int lane = threadIdx.x & 31;
  const int wv   = threadIdx.x >> 5;
  const int hf   = lane >> 4, lm = lane & 15;
  const int m0   = blockIdx.x * 64 + wv * 16;
  const int row  = m0 + lm;
  const v8f vz = {0.f,0.f,0.f,0.f,0.f,0.f,0.f,0.f};
  v8f acc[4] = {vz, vz, vz, vz};

  {
    const int   r1 = idx1 ? idx1[row] : row;
    const float sc = scale1 ? scale1[row] : 1.0f;
    const float* p1 = X1 + (size_t)r1 * 64;
#pragma unroll
    for (int s2 = 0; s2 < 2; ++s2) {
      v16bf a = build_afrag(p1 + s2 * 32 + hf * 8, sc);
      const int s = s2 * 2 + hf;
#pragma unroll
      for (int j = 0; j < 4; ++j) {
        v16bf b = load_bfrag2(W1p + ((((s2 * 2) * 64 + j * 16 + lm) << 4) + hf * 8), W1p + ((((s2 * 2 + 1) * 64 + j * 16 + lm) << 4) + hf * 8));
        acc[j] = wmma_bf16(a, b, acc[j]);
      }
    }
  }
  if (X2) {
    const int r2 = idx2 ? idx2[row] : row;
    const float* p2 = X2 + (size_t)r2 * 64;
#pragma unroll
    for (int s2 = 0; s2 < 2; ++s2) {
      v16bf a = build_afrag(p2 + s2 * 32 + hf * 8, 1.0f);
      const int s = s2 * 2 + hf;
#pragma unroll
      for (int j = 0; j < 4; ++j) {
        v16bf b = load_bfrag2(W2p + ((((s2 * 2) * 64 + j * 16 + lm) << 4) + hf * 8), W2p + ((((s2 * 2 + 1) * 64 + j * 16 + lm) << 4) + hf * 8));
        acc[j] = wmma_bf16(a, b, acc[j]);
      }
    }
  }
#pragma unroll
  for (int j = 0; j < 4; ++j) {
    const float bv = bias[j * 16 + lm];
#pragma unroll
    for (int v = 0; v < 8; ++v) {
      float x = fmaxf(acc[j][v] + bv, 0.0f);
      out[(size_t)(m0 + v + hf * 8) * 64 + j * 16 + lm] = x;
    }
  }
}

// ---------------------------------------------------------------------------
// out[4096 x 64] = Mat[rows[i], :8192] @ Bp.   K split across 8 waves.
// A tiles (16 rows x 32 K, f32) are staged into LDS with double-buffered
// GLOBAL_LOAD_ASYNC_TO_LDS_B128: 4 async b128 loads per tile, lanes covering
// consecutive 128B runs of 4 rows each (fully coalesced).  s_wait_asynccnt
// gates consumption; rows padded to 40 floats so every 32B fragment read is
// aligned and bank-balanced.  Reduction buffer aliases the (dead) stage
// buffer, separated by a workgroup barrier.  grid = 256, block = 256.
// ---------------------------------------------------------------------------
#define STRIDE_F 40            // padded floats per staged row
#define BUF_F    (16 * STRIDE_F)   // floats per stage buffer (640)

__global__ __launch_bounds__(256) void k_big_rowgather(
    const float* __restrict__ Mat, const int* __restrict__ rows,
    const unsigned short* __restrict__ Bp, float* __restrict__ out)
{
  __shared__ __align__(128) float smem[8 * 2 * BUF_F];   // 40960 B
  const int lane = threadIdx.x & 31;
  const int wv   = threadIdx.x >> 5;
  const int hf   = lane >> 4, lm = lane & 15;
  const int i0   = blockIdx.x * 16;

  // 4 row-base pointers per lane: async pass q covers rows q*4 + (lane>>3)
  const float* rb[4];
#pragma unroll
  for (int q = 0; q < 4; ++q)
    rb[q] = Mat + (size_t)rows[i0 + q * 4 + (lane >> 3)] * 8192;
  const int c4 = lane & 7;                       // float4 column within tile
  float* abuf = smem + wv * (2 * BUF_F);
  const unsigned abase =
      (unsigned)(size_t)(smem) + (unsigned)(wv * (2 * BUF_F) * 4);
  const unsigned drow = (unsigned)(((lane >> 3) * STRIDE_F + c4 * 4) * 4);

  const v8f vz = {0.f,0.f,0.f,0.f,0.f,0.f,0.f,0.f};
  v8f acc[4] = {vz, vz, vz, vz};
  const int kbeg = wv * 1024, kend = kbeg + 1024;

  // prologue: stage first tile into buffer 0
#pragma unroll
  for (int q = 0; q < 4; ++q) {
    const float* ga = rb[q] + kbeg + c4 * 4;
    unsigned ld = abase + drow + (unsigned)(q * 4 * STRIDE_F * 4);
    asm volatile("global_load_async_to_lds_b128 %0, %1, off"
                 :: "v"(ld), "v"(ga) : "memory");
  }

  int b = 0;
  for (int k0 = kbeg; k0 < kend; k0 += 32) {
    const int nb = b ^ 1;
    if (k0 + 32 < kend) {                       // stage next tile
#pragma unroll
      for (int q = 0; q < 4; ++q) {
        const float* ga = rb[q] + (k0 + 32) + c4 * 4;
        unsigned ld = abase + (unsigned)(nb * BUF_F * 4) + drow +
                      (unsigned)(q * 4 * STRIDE_F * 4);
        asm volatile("global_load_async_to_lds_b128 %0, %1, off"
                     :: "v"(ld), "v"(ga) : "memory");
      }
      asm volatile("s_wait_asynccnt 0x4" ::: "memory");  // current tile done
    } else {
      asm volatile("s_wait_asynccnt 0x0" ::: "memory");
    }
    // consume buffer b
    const float* lp = abuf + b * BUF_F + lm * STRIDE_F + hf * 8;
    v16bf a = build_afrag(lp, 1.0f);
    const int s = (k0 >> 4) + hf;
#pragma unroll
    for (int j = 0; j < 4; ++j) {
      v16bf bb = load_bfrag2(Bp + ((((k0 >> 4) * 64 + j * 16 + lm) << 4) + hf * 8), Bp + ((((k0 >> 4) + 1) * 64 + j * 16 + lm) << 4) + hf * 8);
      acc[j] = wmma_bf16(a, bb, acc[j]);
    }
    b = nb;
  }

  __syncthreads();                       // stage buffers dead; alias as red[]
  float* red = smem;                     // [8][16][64]
#pragma unroll
  for (int j = 0; j < 4; ++j)
#pragma unroll
    for (int v = 0; v < 8; ++v)
      red[(wv * 16 + v + hf * 8) * 64 + j * 16 + lm] = acc[j][v];
  __syncthreads();
  for (int e = threadIdx.x; e < 1024; e += 256) {
    const int rr = e >> 6, cc = e & 63;
    float s = 0.f;
#pragma unroll
    for (int w = 0; w < 8; ++w) s += red[(w * 16 + rr) * 64 + cc];
    out[(size_t)(i0 + rr) * 64 + cc] = s;
  }
}

// ---------------------------------------------------------------------------
// Ct[64 x 8192] = Uh^T @ R.  A fragments come straight from the packed Uh
// (B-pack layout == A^T layout).  B fragments stream rows of R coalesced.
// grid = 8192/32 = 256, block = 128 (4 waves split K=8192).
// ---------------------------------------------------------------------------
__global__ __launch_bounds__(128) void k_big_transposed(
    const unsigned short* __restrict__ Ap, const float* __restrict__ Rm,
    float* __restrict__ Ct)
{
  __shared__ float red[4][64][32];
  const int lane = threadIdx.x & 31;
  const int wv   = threadIdx.x >> 5;
  const int hf   = lane >> 4, lm = lane & 15;
  const int n0   = blockIdx.x * 32;
  const v8f vz = {0.f,0.f,0.f,0.f,0.f,0.f,0.f,0.f};
  v8f acc[8] = {vz, vz, vz, vz, vz, vz, vz, vz};

  const int kend = wv * 2048 + 2048;
  for (int k0 = wv * 2048; k0 < kend; k0 += 32) {
    if ((k0 & 127) == 0)   // prefetch rows 128 K ahead (one line per lane)
      __builtin_prefetch(Rm + (size_t)(k0 + 128 + lane) * 8192 + n0, 0, 1);
    const int s0 = k0 >> 4;
    v16bf a[4];
#pragma unroll
    for (int mt = 0; mt < 4; ++mt) {
      const unsigned short* pl = Ap + (((s0    ) * 64 + mt * 16 + lm) << 4) + hf * 8;
      const unsigned short* ph = Ap + (((s0 + 1) * 64 + mt * 16 + lm) << 4) + hf * 8;
      a[mt] = load_bfrag2(pl, ph);
    }
#pragma unroll
    for (int jt = 0; jt < 2; ++jt) {
      const size_t n = (size_t)(n0 + jt * 16 + lm);
      v16bf b;
#pragma unroll
      for (int j = 0; j < 16; ++j)
        b[j] = (__bf16)(Rm[(size_t)(k0 + hf * 8 + (j & 7) + (j & 8) * 2) * 8192 + n]);
#pragma unroll
      for (int mt = 0; mt < 4; ++mt)
        acc[mt * 2 + jt] = wmma_bf16(a[mt], b, acc[mt * 2 + jt]);
    }
  }
#pragma unroll
  for (int mt = 0; mt < 4; ++mt)
#pragma unroll
    for (int jt = 0; jt < 2; ++jt)
#pragma unroll
      for (int v = 0; v < 8; ++v)
        red[wv][mt * 16 + v + hf * 8][jt * 16 + lm] = acc[mt * 2 + jt][v];
  __syncthreads();
  for (int e = threadIdx.x; e < 2048; e += 128) {
    const int m = e >> 5, c = e & 31;
    float s = 0.f;
#pragma unroll
    for (int w = 0; w < 4; ++w) s += red[w][m][c];
    Ct[(size_t)m * 8192 + n0 + c] = s;
  }
}

// ---- Pack f32 [K x 64] row-major into bf16 B-fragment layout --------------
__global__ __launch_bounds__(256) void k_pack_b(
    const float* __restrict__ src, unsigned short* __restrict__ dst, int K)
{
  const int tid = blockIdx.x * 256 + threadIdx.x;
  const int total = (K >> 4) * 64;
  if (tid >= total) return;
  const int s = tid >> 6, n = tid & 63;
  v16bf t;
#pragma unroll
  for (int j = 0; j < 16; ++j)
    t[j] = (__bf16)(src[(size_t)(s * 16 + j) * 64 + n]);
  *(v16bf*)(dst + ((size_t)tid << 4)) = t;
}

// ---- e_r = [1..5]@w_r + b_r ; cu0 = e_r@Wu0[64:]+bu0 ; cv0 likewise -------
__global__ void k_prep(const float* __restrict__ w_r, const float* __restrict__ b_r,
                       const float* __restrict__ Wu0, const float* __restrict__ bu0,
                       const float* __restrict__ Wv0, const float* __restrict__ bv0,
                       float* __restrict__ cu0, float* __restrict__ cv0)
{
  __shared__ float er[64];
  const int d = threadIdx.x;
  float e = b_r[d];
#pragma unroll
  for (int s = 0; s < 5; ++s) e += (float)(s + 1) * w_r[s * 64 + d];
  er[d] = e;
  __syncthreads();
  float a = bu0[d], b = bv0[d];
  for (int k = 0; k < 64; ++k) {
    const float ek = er[k];
    a += ek * Wu0[(64 + k) * 64 + d];
    b += ek * Wv0[(64 + k) * 64 + d];
  }
  cu0[d] = a; cv0[d] = b;
}

// ---- scores[i] = H[i,:] . w2 + b2 -----------------------------------------
__global__ __launch_bounds__(256) void k_rowdot(
    const float* __restrict__ H, const float* __restrict__ w2,
    const float* __restrict__ b2, float* __restrict__ sc)
{
  const int i = blockIdx.x * 256 + threadIdx.x;
  const float* h = H + (size_t)i * 64;
  float s = b2[0];
#pragma unroll
  for (int n = 0; n < 64; ++n) s += h[n] * w2[n];
  sc[i] = s;
}

// ---- softmax over 4096 elements (single block, deterministic) -------------
__global__ __launch_bounds__(256) void k_softmax4096(
    const float* __restrict__ sc, float* __restrict__ pr)
{
  __shared__ float red[256];
  const int t = threadIdx.x;
  float m = -3.402823466e38f;
  for (int i = t; i < 4096; i += 256) m = fmaxf(m, sc[i]);
  red[t] = m; __syncthreads();
  for (int o = 128; o > 0; o >>= 1) { if (t < o) red[t] = fmaxf(red[t], red[t + o]); __syncthreads(); }
  const float mx = red[0];
  __syncthreads();
  float s = 0.f;
  for (int i = t; i < 4096; i += 256) s += __expf(sc[i] - mx);
  red[t] = s; __syncthreads();
  for (int o = 128; o > 0; o >>= 1) { if (t < o) red[t] += red[t + o]; __syncthreads(); }
  const float inv = 1.0f / red[0];
  for (int i = t; i < 4096; i += 256) pr[i] = __expf(sc[i] - mx) * inv;
}

// ---- Tm[i][d] = Ct[d][item[i]] --------------------------------------------
__global__ __launch_bounds__(256) void k_gather_cols(
    const float* __restrict__ Ct, const int* __restrict__ items,
    float* __restrict__ Tm)
{
  const int tid = blockIdx.x * 256 + threadIdx.x;
  const int i = tid >> 6, d = tid & 63;
  Tm[tid] = Ct[(size_t)d * 8192 + items[i]];
}

// ===========================================================================
extern "C" void kernel_launch(void* const* d_in, const int* in_sizes, int n_in,
                              void* d_out, int out_size, void* d_ws, size_t ws_size,
                              hipStream_t stream) {
  (void)in_sizes; (void)n_in; (void)out_size; (void)ws_size;
  typedef unsigned short u16;

  const int*   busers = (const int*)  d_in[0];
  const int*   bitems = (const int*)  d_in[1];
  const float* R      = (const float*)d_in[2];
  const float* S      = (const float*)d_in[3];
  const float* U      = (const float*)d_in[4];
  const float* V      = (const float*)d_in[5];
  const float* w_r  = (const float*)d_in[6];  const float* b_r  = (const float*)d_in[7];
  const float* Wu0  = (const float*)d_in[8];  const float* bu0  = (const float*)d_in[9];
  const float* Wu1  = (const float*)d_in[10]; const float* bu1  = (const float*)d_in[11];
  const float* Wv0  = (const float*)d_in[12]; const float* bv0  = (const float*)d_in[13];
  const float* Wv1  = (const float*)d_in[14]; const float* bv1  = (const float*)d_in[15];
  const float* Wc0  = (const float*)d_in[16]; const float* bc0  = (const float*)d_in[17];
  const float* Wc1  = (const float*)d_in[18]; const float* bc1  = (const float*)d_in[19];
  const float* Wg0  = (const float*)d_in[20]; const float* bg0  = (const float*)d_in[21];
  const float* Wg1  = (const float*)d_in[22]; const float* bg1  = (const float*)d_in[23];
  const float* W_I  = (const float*)d_in[24]; const float* b_I  = (const float*)d_in[25];
  const float* W_Sm = (const float*)d_in[26]; const float* b_Sm = (const float*)d_in[27];
  const float* W1A  = (const float*)d_in[28]; const float* b1A  = (const float*)d_in[29];
  const float* w2A  = (const float*)d_in[30]; const float* b2A  = (const float*)d_in[31];
  const float* W1B  = (const float*)d_in[32]; const float* b1B  = (const float*)d_in[33];
  const float* w2B  = (const float*)d_in[34]; const float* b2B  = (const float*)d_in[35];
  const float* W1M  = (const float*)d_in[36]; const float* b1M  = (const float*)d_in[37];
  const float* w2M  = (const float*)d_in[38]; const float* b2M  = (const float*)d_in[39];
  const float* W_Z  = (const float*)d_in[40]; const float* b_Z  = (const float*)d_in[41];
  float* outG = (float*)d_out;

  // ---- workspace carve-out (256B aligned) ---------------------------------
  char* base = (char*)d_ws; size_t off = 0;
  auto alloc = [&](size_t bytes) -> void* {
    void* p = base + off; off = (off + bytes + 255) & ~(size_t)255; return p;
  };
  const size_t WPK = 64 * 64 * sizeof(u16);          // one packed 64x64 weight
  u16* pWu0t=(u16*)alloc(WPK); u16* pWu1=(u16*)alloc(WPK);
  u16* pWv0t=(u16*)alloc(WPK); u16* pWv1=(u16*)alloc(WPK);
  u16* pA1=(u16*)alloc(WPK);   u16* pA2=(u16*)alloc(WPK);
  u16* pB1=(u16*)alloc(WPK);   u16* pB2=(u16*)alloc(WPK);
  u16* pM1=(u16*)alloc(WPK);   u16* pM2=(u16*)alloc(WPK);
  u16* pWI=(u16*)alloc(WPK);   u16* pWSm=(u16*)alloc(WPK); u16* pWZ=(u16*)alloc(WPK);
  u16* pC0a=(u16*)alloc(WPK);  u16* pC0b=(u16*)alloc(WPK); u16* pC1=(u16*)alloc(WPK);
  u16* pG0a=(u16*)alloc(WPK);  u16* pG0b=(u16*)alloc(WPK); u16* pG1=(u16*)alloc(WPK);
  u16* pUh = (u16*)alloc((size_t)8192 * 64 * sizeof(u16));
  u16* pVh = (u16*)alloc((size_t)8192 * 64 * sizeof(u16));
  float* cu0 = (float*)alloc(64 * 4);
  float* cv0 = (float*)alloc(64 * 4);
  float* h0  = (float*)alloc((size_t)8192 * 64 * 4);
  float* Uh  = (float*)alloc((size_t)8192 * 64 * 4);
  float* Vh  = (float*)alloc((size_t)8192 * 64 * 4);
  float* P   = (float*)alloc((size_t)4096 * 64 * 4);
  float* Q   = (float*)alloc((size_t)4096 * 64 * 4);
  float* Ct  = (float*)alloc((size_t)64 * 8192 * 4);
  float* Tm  = (float*)alloc((size_t)4096 * 64 * 4);
  float* hid = (float*)alloc((size_t)4096 * 64 * 4);
  float* scb = (float*)alloc(4096 * 4);
  float* prb = (float*)alloc(4096 * 4);
  float* HI  = (float*)alloc((size_t)4096 * 64 * 4);
  float* HS  = (float*)alloc((size_t)4096 * 64 * 4);
  float* Hc0 = (float*)alloc((size_t)4096 * 64 * 4);
  float* Hc  = (float*)alloc((size_t)4096 * 64 * 4);
  float* Z   = (float*)alloc((size_t)4096 * 64 * 4);
  float* G0  = (float*)alloc((size_t)4096 * 64 * 4);

  const float* FN = nullptr; const int* IN_ = nullptr; const u16* UN = nullptr;

  // ---- 1. prep + weight packing -------------------------------------------
  k_prep<<<1, 64, 0, stream>>>(w_r, b_r, Wu0, bu0, Wv0, bv0, cu0, cv0);
  k_pack_b<<<1,256,0,stream>>>(Wu0,        pWu0t, 64);
  k_pack_b<<<1,256,0,stream>>>(Wu1,        pWu1,  64);
  k_pack_b<<<1,256,0,stream>>>(Wv0,        pWv0t, 64);
  k_pack_b<<<1,256,0,stream>>>(Wv1,        pWv1,  64);
  k_pack_b<<<1,256,0,stream>>>(W1A,        pA1,   64);
  k_pack_b<<<1,256,0,stream>>>(W1A + 4096, pA2,   64);
  k_pack_b<<<1,256,0,stream>>>(W1B,        pB1,   64);
  k_pack_b<<<1,256,0,stream>>>(W1B + 4096, pB2,   64);
  k_pack_b<<<1,256,0,stream>>>(W1M,        pM1,   64);
  k_pack_b<<<1,256,0,stream>>>(W1M + 4096, pM2,   64);
  k_pack_b<<<1,256,0,stream>>>(W_I,        pWI,   64);
  k_pack_b<<<1,256,0,stream>>>(W_Sm,       pWSm,  64);
  k_pack_b<<<1,256,0,stream>>>(W_Z,        pWZ,   64);
  k_pack_b<<<1,256,0,stream>>>(Wc0,        pC0a,  64);
  k_pack_b<<<1,256,0,stream>>>(Wc0 + 4096, pC0b,  64);
  k_pack_b<<<1,256,0,stream>>>(Wc1,        pC1,   64);
  k_pack_b<<<1,256,0,stream>>>(Wg0,        pG0a,  64);
  k_pack_b<<<1,256,0,stream>>>(Wg0 + 4096, pG0b,  64);
  k_pack_b<<<1,256,0,stream>>>(Wg1,        pG1,   64);

  // ---- 2. node MLPs: Uh, Vh (e_r concat folded into cu0/cv0) --------------
  k_small_gemm<<<128,128,0,stream>>>(U,  IN_, FN, pWu0t, FN, IN_, UN, cu0, h0);
  k_small_gemm<<<128,128,0,stream>>>(h0, IN_, FN, pWu1,  FN, IN_, UN, bu1, Uh);
  k_small_gemm<<<128,128,0,stream>>>(V,  IN_, FN, pWv0t, FN, IN_, UN, cv0, h0);
  k_small_gemm<<<128,128,0,stream>>>(h0, IN_, FN, pWv1,  FN, IN_, UN, bv1, Vh);
  k_pack_b<<<128,256,0,stream>>>(Uh, pUh, 8192);
  k_pack_b<<<128,256,0,stream>>>(Vh, pVh, 8192);

  // ---- 3. big GEMMs (attention-independent; scales commute) ---------------
  k_big_rowgather <<<256,256,0,stream>>>(R, busers, pVh, P);   // R[user] @ Vh
  k_big_rowgather <<<256,256,0,stream>>>(S, busers, pUh, Q);   // S[user] @ Uh
  k_big_transposed<<<256,128,0,stream>>>(pUh, R, Ct);          // Uh^T @ R

  // ---- 4. attention A -> H_I ----------------------------------------------
  k_small_gemm<<<64,128,0,stream>>>(Vh, bitems, FN, pA1, Uh, busers, pA2, b1A, hid);
  k_rowdot<<<16,256,0,stream>>>(hid, w2A, b2A, scb);
  k_softmax4096<<<1,256,0,stream>>>(scb, prb);
  k_small_gemm<<<64,128,0,stream>>>(P, IN_, prb, pWI, FN, IN_, UN, b_I, HI);

  // ---- 5. attention B -> H_S ----------------------------------------------
  k_small_gemm<<<64,128,0,stream>>>(HI, IN_, FN, pB1, Uh, busers, pB2, b1B, hid);
  k_rowdot<<<16,256,0,stream>>>(hid, w2B, b2B, scb);
  k_softmax4096<<<1,256,0,stream>>>(scb, prb);
  k_small_gemm<<<64,128,0,stream>>>(Q, IN_, prb, pWSm, FN, IN_, UN, b_Sm, HS);

  // ---- 6. H = MLP([H_I, H_S]) ---------------------------------------------
  k_small_gemm<<<64,128,0,stream>>>(HI,  IN_, FN, pC0a, HS, IN_, pC0b, bc0, Hc0);
  k_small_gemm<<<64,128,0,stream>>>(Hc0, IN_, FN, pC1,  FN, IN_, UN,   bc1, Hc);

  // ---- 7. attention M -> Z ------------------------------------------------
  k_small_gemm<<<64,128,0,stream>>>(Uh, busers, FN, pM1, Vh, bitems, pM2, b1M, hid);
  k_rowdot<<<16,256,0,stream>>>(hid, w2M, b2M, scb);
  k_softmax4096<<<1,256,0,stream>>>(scb, prb);
  k_gather_cols<<<1024,256,0,stream>>>(Ct, bitems, Tm);
  k_small_gemm<<<64,128,0,stream>>>(Tm, IN_, prb, pWZ, FN, IN_, UN, b_Z, Z);

  // ---- 8. G = MLP([H, Z]) -> d_out ----------------------------------------
  k_small_gemm<<<64,128,0,stream>>>(Hc, IN_, FN, pG0a, Z, IN_, pG0b, bg0, G0);
  k_small_gemm<<<64,128,0,stream>>>(G0, IN_, FN, pG1,  FN, IN_, UN,  bg1, outG);
}
